// Encoder_28767690948836
// MI455X (gfx1250) — compile-verified, matrix operand order checked by reading, not run on hardware
//
#include <hip/hip_runtime.h>

// ---------------- constants ----------------
#define BATCH 128
#define SEQT  256
#define EMB   256
#define UNITS 1024
#define N3    3072   // 3*UNITS

typedef __attribute__((ext_vector_type(16))) __bf16 v16bf;
typedef __attribute__((ext_vector_type(8)))  __bf16 v8bf;
typedef __attribute__((ext_vector_type(8)))  float  v8f;

// f32 -> bf16 round-to-nearest-even (bit pattern)
__device__ __forceinline__ unsigned short f2bf(float f) {
    unsigned int u = __float_as_uint(f);
    unsigned int r = u + 0x7FFFu + ((u >> 16) & 1u);
    return (unsigned short)(r >> 16);
}

// WMMA bf16 tile index math (matches CDNA5 ISA 16-bit A/B layouts):
//   lane l, element e (0..15):  K = kbase + ((l>=16)?8:0) + (e<8 ? e : e+8)
//   A: row M = l%16      B: col N = l%16

// ---- pack W [Kdim, 3072] f32 (both dirs) into B-tiles [dir][192][KT][512] bf16 ----
__global__ void pack_w(const float* __restrict__ Wf, const float* __restrict__ Wb,
                       unsigned short* __restrict__ outp, int KT) {
    long long idx = (long long)blockIdx.x * 256 + threadIdx.x;
    long long total = 2LL * 192 * KT * 512;
    if (idx >= total) return;
    int e    = (int)(idx & 15);
    int lane = (int)((idx >> 4) & 31);
    long long rest = idx >> 9;
    int kt = (int)(rest % KT); rest /= KT;
    int nt = (int)(rest % 192);
    int dir = (int)(rest / 192);
    int K = kt * 32 + ((lane >> 4) << 3) + (e < 8 ? e : e + 8);
    int N = nt * 16 + (lane & 15);
    const float* W = dir ? Wb : Wf;
    outp[idx] = f2bf(W[(long long)K * N3 + N]);
}

// ---- gather embedding + pack into A-tiles [T][8 mt][8 kt][512] bf16 ----
__global__ void pack_emb(const int* __restrict__ seq, const float* __restrict__ tab,
                         unsigned short* __restrict__ outp) {
    long long idx = (long long)blockIdx.x * 256 + threadIdx.x;   // 256*8*8*512 total
    int e    = (int)(idx & 15);
    int lane = (int)((idx >> 4) & 31);
    int kt   = (int)((idx >> 9) & 7);
    int mt   = (int)((idx >> 12) & 7);
    int t    = (int)(idx >> 15);
    int b = mt * 16 + (lane & 15);
    int K = kt * 32 + ((lane >> 4) << 3) + (e < 8 ? e : e + 8);
    int tok = seq[b * SEQT + t];
    outp[idx] = f2bf(tab[(long long)tok * EMB + K]);
}

// ---- initialize ping-pong state (buffer 0) for both directions ----
__global__ void init_h(const float* __restrict__ h0, float* __restrict__ hf,
                       unsigned short* __restrict__ hb) {
    int idx = blockIdx.x * 256 + threadIdx.x;          // 2*128*1024 total
    int dir = idx / (BATCH * UNITS);
    int i   = idx % (BATCH * UNITS);
    float v = h0[i];
    hf[(dir * 2 + 0) * (BATCH * UNITS) + i] = v;
    hb[(dir * 2 + 0) * (BATCH * UNITS) + i] = f2bf(v);
}

// ---- one GRU time step, both directions (gridDim.y), 512 waves of WMMA ----
__global__ void __launch_bounds__(256)
gru_step(const unsigned short* __restrict__ Wr_p,   // [2][192][32][512]
         const unsigned short* __restrict__ Wk_p,   // [2][192][8][512]
         const unsigned short* __restrict__ emb_p,  // [T][8][8][512]
         const float* __restrict__ bias_f, const float* __restrict__ bias_b,
         const int* __restrict__ seq,
         float* __restrict__ h_f32, unsigned short* __restrict__ h_bf16,
         float* __restrict__ out, float* __restrict__ hlast, int s) {
    const int dir  = blockIdx.y;
    const int wave = threadIdx.x >> 5;
    const int lane = threadIdx.x & 31;
    const int w    = blockIdx.x * 8 + wave;  // 0..511
    const int mt   = w >> 6;                 // 0..7   (batch tiles)
    const int nt   = w & 63;                 // 0..63  (unit tiles)
    const int t_eff = dir ? (SEQT - 1 - s) : s;
    const int inb = s & 1, outb = inb ^ 1;
    const int HS = BATCH * UNITS;
    const float*          hin   = h_f32  + (dir * 2 + inb)  * HS;
    float*                hout  = h_f32  + (dir * 2 + outb) * HS;
    const unsigned short* hbin  = h_bf16 + (dir * 2 + inb)  * HS;
    unsigned short*       hbout = h_bf16 + (dir * 2 + outb) * HS;
    const float*          bias  = dir ? bias_b : bias_f;
    const unsigned short* Wr_d  = Wr_p + (long long)dir * 192 * 32 * 512;
    const unsigned short* Wk_d  = Wk_p + (long long)dir * 192 * 8 * 512;

    const int l16   = lane & 15;
    const int khalf = (lane >> 4) << 3;      // 0 or 8
    const int arow  = mt * 16 + l16;         // A-matrix row = batch index

    v8f acc_z  = {};
    v8f acc_r  = {};
    v8f acc_xh = {};
    v8f acc_rh = {};

    // ---- recurrent part: K = 1024 (32 k-tiles), B from packed Wr ----
    #pragma unroll 4
    for (int kt = 0; kt < 32; ++kt) {
        const unsigned short* hrow = hbin + arow * UNITS + kt * 32 + khalf;
        v8bf lo = *reinterpret_cast<const v8bf*>(hrow);
        v8bf hi = *reinterpret_cast<const v8bf*>(hrow + 16);
        v16bf a = __builtin_shufflevector(lo, hi, 0,1,2,3,4,5,6,7,8,9,10,11,12,13,14,15);
        v16bf bz = *reinterpret_cast<const v16bf*>(Wr_d + (((nt      ) * 32 + kt) * 512) + lane * 16);
        v16bf br = *reinterpret_cast<const v16bf*>(Wr_d + (((nt +  64) * 32 + kt) * 512) + lane * 16);
        v16bf bh = *reinterpret_cast<const v16bf*>(Wr_d + (((nt + 128) * 32 + kt) * 512) + lane * 16);
        acc_z  = __builtin_amdgcn_wmma_f32_16x16x32_bf16(false, a, false, bz, (short)0, acc_z,  false, false);
        acc_r  = __builtin_amdgcn_wmma_f32_16x16x32_bf16(false, a, false, br, (short)0, acc_r,  false, false);
        acc_rh = __builtin_amdgcn_wmma_f32_16x16x32_bf16(false, a, false, bh, (short)0, acc_rh, false, false);
    }

    // ---- input-projection part: K = 256 (8 k-tiles), A from packed emb ----
    #pragma unroll 4
    for (int kt = 0; kt < 8; ++kt) {
        v16bf a  = *reinterpret_cast<const v16bf*>(emb_p + ((((long long)t_eff * 8 + mt) * 8 + kt) * 512) + lane * 16);
        v16bf bz = *reinterpret_cast<const v16bf*>(Wk_d + (((nt      ) * 8 + kt) * 512) + lane * 16);
        v16bf br = *reinterpret_cast<const v16bf*>(Wk_d + (((nt +  64) * 8 + kt) * 512) + lane * 16);
        v16bf bh = *reinterpret_cast<const v16bf*>(Wk_d + (((nt + 128) * 8 + kt) * 512) + lane * 16);
        acc_z  = __builtin_amdgcn_wmma_f32_16x16x32_bf16(false, a, false, bz, (short)0, acc_z,  false, false);
        acc_r  = __builtin_amdgcn_wmma_f32_16x16x32_bf16(false, a, false, br, (short)0, acc_r,  false, false);
        acc_xh = __builtin_amdgcn_wmma_f32_16x16x32_bf16(false, a, false, bh, (short)0, acc_xh, false, false);
    }

    // ---- epilogue: gates + mask-carry + stores (C/D layout: N=lane%16, M=i+8*(lane>=16)) ----
    const int n = nt * 16 + l16;
    const float bzc = bias[n]             + bias[N3 + n];
    const float brc = bias[UNITS + n]     + bias[N3 + UNITS + n];
    const float bxh = bias[2 * UNITS + n];
    const float brh = bias[N3 + 2 * UNITS + n];
    const int rbase = mt * 16 + ((lane >> 4) << 3);
    #pragma unroll
    for (int i = 0; i < 8; ++i) {
        int row = rbase + i;
        float z  = 1.0f / (1.0f + __expf(-(acc_z[i] + bzc)));
        float rg = 1.0f / (1.0f + __expf(-(acc_r[i] + brc)));
        float hh = tanhf(acc_xh[i] + bxh + rg * (acc_rh[i] + brh));
        float ho = hin[row * UNITS + n];
        float hn = z * ho + (1.0f - z) * hh;
        bool  m  = seq[row * SEQT + t_eff] != 0;
        float hc = m ? hn : ho;
        hout[row * UNITS + n]  = hc;
        hbout[row * UNITS + n] = f2bf(hc);
        // output[b, s, dir*U + n]  (bwd kept in reversed order, matching reference concat)
        out[((long long)row * SEQT + s) * (2 * UNITS) + dir * UNITS + n] = hc;
        if (dir == 1 && s == SEQT - 1) hlast[row * UNITS + n] = hc;
    }
}

extern "C" void kernel_launch(void* const* d_in, const int* in_sizes, int n_in,
                              void* d_out, int out_size, void* d_ws, size_t ws_size,
                              hipStream_t stream) {
    const int*   seq    = (const int*)d_in[0];
    const float* hidden = (const float*)d_in[1];
    const float* etab   = (const float*)d_in[2];
    const float* Wk_f   = (const float*)d_in[3];
    const float* Wr_f   = (const float*)d_in[4];
    const float* b_f    = (const float*)d_in[5];
    const float* Wk_b   = (const float*)d_in[6];
    const float* Wr_b   = (const float*)d_in[7];
    const float* b_b    = (const float*)d_in[8];

    char* ws = (char*)d_ws;
    size_t oWr  = 0;                                   // 2*192*32*512 bf16 = 12.6 MB
    size_t oWk  = oWr  + 2ULL * 192 * 32 * 512 * 2;    // 2*192*8*512 bf16  =  3.1 MB
    size_t oEmb = oWk  + 2ULL * 192 * 8 * 512 * 2;     // 256*8*8*512 bf16  = 16.8 MB
    size_t oHf  = oEmb + 256ULL * 64 * 512 * 2;        // 2 dir * 2 buf * 128K f32 = 2 MB
    size_t oHb  = oHf  + 2ULL * 2 * BATCH * UNITS * 4; // bf16 mirror = 1 MB
    unsigned short* Wr_p  = (unsigned short*)(ws + oWr);
    unsigned short* Wk_p  = (unsigned short*)(ws + oWk);
    unsigned short* emb_p = (unsigned short*)(ws + oEmb);
    float*          h_f   = (float*)(ws + oHf);
    unsigned short* h_b   = (unsigned short*)(ws + oHb);

    float* out   = (float*)d_out;
    float* hlast = out + (long long)BATCH * SEQT * 2 * UNITS;

    pack_w  <<<(2 * 192 * 32 * 512) / 256, 256, 0, stream>>>(Wr_f, Wr_b, Wr_p, 32);
    pack_w  <<<(2 * 192 *  8 * 512) / 256, 256, 0, stream>>>(Wk_f, Wk_b, Wk_p, 8);
    pack_emb<<<(SEQT * 8 * 8 * 512) / 256, 256, 0, stream>>>(seq, etab, emb_p);
    init_h  <<<(2 * BATCH * UNITS) / 256, 256, 0, stream>>>(hidden, h_f, h_b);

    for (int s = 0; s < SEQT; ++s)
        gru_step<<<dim3(64, 2), 256, 0, stream>>>(Wr_p, Wk_p, emb_p, b_f, b_b, seq,
                                                  h_f, h_b, out, hlast, s);
}
